// PyGHeteroEncoder_50903952392275
// MI455X (gfx1250) — compile-verified, matrix operand order checked by reading, not run on hardware
//
#include <hip/hip_runtime.h>
#include <cstdint>
#include <cstddef>

// ---------------------------------------------------------------------------
// Problem constants (match reference)
// ---------------------------------------------------------------------------
#define NG 50000
#define NP 20000
#define NV 10000
#define NT (NG + NP + NV)
#define HID 128
#define EDG 250000
#define ALPHA_F 0.1f
#define K_APPNP 8

// ---------------------------------------------------------------------------
// CDNA5 WMMA types
// ---------------------------------------------------------------------------
typedef __attribute__((ext_vector_type(16))) __bf16 v16bf;
typedef __attribute__((ext_vector_type(8)))  float  v8f;

union BF16Frag {
    v16bf v;
    uint4 q[2];
    unsigned short u[16];
};

// float -> bf16 (round to nearest even), pure bit manipulation
__device__ __forceinline__ unsigned short f2bf(float f) {
    unsigned int u = __float_as_uint(f);
    unsigned int r = u + 0x7FFFu + ((u >> 16) & 1u);
    return (unsigned short)(r >> 16);
}

// ---------------------------------------------------------------------------
// WMMA GEMM:  Y[r,n] = relu?( X[r,:]·W[n,:] + bias[n] (+ Y[r,n] if ACCUM) )
//   X: [nrows, KDIM] f32 row-major, W: [128, KDIM] f32 row-major
// Block = 128 threads (4 wave32s); 64 rows per block, wave w owns rows
// [base+16w, base+16w+16). Each wave holds 8 v8f accumulators covering all
// 128 output channels. K loop in 32-steps; A and W staged in LDS as bf16 in
// WMMA fragment order (ISA 7.12.2): for a 16x32 16-bit tile, element (m,k)
// lives at lane = m + 16*k[3], elem = k[2:0] | (k[4]<<3). Each lane then
// loads its 32-byte fragment contiguously (ds_load_b128 x2).
// ---------------------------------------------------------------------------
template <int KDIM, bool BIAS, bool RELU, bool ACCUM>
__global__ __launch_bounds__(128) void gemm_nt_wmma(
    const float* __restrict__ X, const float* __restrict__ W,
    const float* __restrict__ bias, float* __restrict__ Y, int nrows) {
    constexpr int KT = KDIM / 32;

    __shared__ __attribute__((aligned(16))) unsigned short Ws[8][32][16]; // 8 n-tiles x frag
    __shared__ __attribute__((aligned(16))) unsigned short As[4][32][16]; // 4 waves  x frag

    const int tid  = threadIdx.x;
    const int wave = tid >> 5;
    const int lane = tid & 31;
    const int rowBase = blockIdx.x * 64;

    v8f acc[8];
    const v8f vzero = {0.f, 0.f, 0.f, 0.f, 0.f, 0.f, 0.f, 0.f};
    for (int i = 0; i < 8; ++i) acc[i] = vzero;

    for (int kt = 0; kt < KT; ++kt) {
        __syncthreads(); // previous iteration's reads complete before overwrite
        // Stage W tile: 128 out-channels x 32 k  (4096 elems, 32/thread)
        for (int i = tid; i < 4096; i += 128) {
            int n = i >> 5;
            int k = i & 31;
            float w = W[(size_t)n * KDIM + kt * 32 + k];
            int hi = (k >> 3) & 1;
            int e  = (k & 7) | (((k >> 4) & 1) << 3);
            Ws[n >> 4][(n & 15) + hi * 16][e] = f2bf(w);
        }
        // Stage A tile: 64 rows x 32 k (2048 elems, 16/thread), zero-pad tail rows
        for (int i = tid; i < 2048; i += 128) {
            int r = i >> 5;
            int k = i & 31;
            int row = rowBase + r;
            float x = (row < nrows) ? X[(size_t)row * KDIM + kt * 32 + k] : 0.0f;
            int hi = (k >> 3) & 1;
            int e  = (k & 7) | (((k >> 4) & 1) << 3);
            As[r >> 4][(r & 15) + hi * 16][e] = f2bf(x);
        }
        __syncthreads();

        BF16Frag a;
        const uint4* ap = (const uint4*)&As[wave][lane][0];
        a.q[0] = ap[0];
        a.q[1] = ap[1];
        for (int nt = 0; nt < 8; ++nt) {
            BF16Frag b;
            const uint4* bp = (const uint4*)&Ws[nt][lane][0];
            b.q[0] = bp[0];
            b.q[1] = bp[1];
            acc[nt] = __builtin_amdgcn_wmma_f32_16x16x32_bf16(
                false, a.v, false, b.v, (short)0, acc[nt], false, false);
        }
    }

    // Store: C/D layout -> n = lane&15, m = j + 8*(lane>=16)
    const int nloc = lane & 15;
    const int mAdd = (lane >> 4) * 8;
    const int row0 = rowBase + wave * 16;
    for (int nt = 0; nt < 8; ++nt) {
        for (int j = 0; j < 8; ++j) {
            int row = row0 + j + mAdd;
            if (row < nrows) {
                int n = nt * 16 + nloc;
                float v = acc[nt][j];
                if (BIAS) v += bias[n];
                float* p = Y + (size_t)row * HID + n;
                if (ACCUM) v += *p;
                if (RELU) v = fmaxf(v, 0.0f);
                *p = v;
            }
        }
    }
}

// ---------------------------------------------------------------------------
// Elementwise / scatter kernels (L2-resident atomic traffic)
// ---------------------------------------------------------------------------
__global__ void fill_k(float* __restrict__ p, float v, long n) {
    long i = blockIdx.x * (long)blockDim.x + threadIdx.x;
    if (i < n) p[i] = v;
}

// out[dst[e], c] += feat[src[e], c]   (float4 gather, scalar f32 atomics)
__global__ void scatter_feat(const float* __restrict__ feat,
                             const int* __restrict__ src,
                             const int* __restrict__ dst,
                             float* __restrict__ out, int E) {
    long tid = blockIdx.x * (long)blockDim.x + threadIdx.x;
    if (tid >= (long)E * 32) return;
    int e  = (int)(tid >> 5);
    int c0 = (int)(tid & 31) << 2;
    int s = src[e], d = dst[e];
    const float4 v = *(const float4*)(feat + (size_t)s * HID + c0);
    float* o = out + (size_t)d * HID + c0;
    atomicAdd(o + 0, v.x);
    atomicAdd(o + 1, v.y);
    atomicAdd(o + 2, v.z);
    atomicAdd(o + 3, v.w);
}

__global__ void count_edges(const int* __restrict__ dst, float* __restrict__ cnt,
                            int E, int off) {
    long t = blockIdx.x * (long)blockDim.x + threadIdx.x;
    if (t < E) atomicAdd(&cnt[dst[t] + off], 1.0f);
}

__global__ void mean_div(float* __restrict__ agg, const float* __restrict__ cnt, long n) {
    long i = blockIdx.x * (long)blockDim.x + threadIdx.x;
    if (i < n) agg[i] /= fmaxf(cnt[i >> 7], 1.0f);
}

// h = relu(h + o*scale)
__global__ void combine_scale_relu(float* __restrict__ h, const float* __restrict__ o,
                                   float scale, long n) {
    long i = blockIdx.x * (long)blockDim.x + threadIdx.x;
    if (i < n) h[i] = fmaxf(h[i] + o[i] * scale, 0.0f);
}

__global__ void relu_ip(float* __restrict__ h, long n) {
    long i = blockIdx.x * (long)blockDim.x + threadIdx.x;
    if (i < n) h[i] = fmaxf(h[i], 0.0f);
}

__global__ void rsqrt_k(float* __restrict__ dinv, const float* __restrict__ deg, int n) {
    long i = blockIdx.x * (long)blockDim.x + threadIdx.x;
    if (i < n) dinv[i] = rsqrtf(deg[i]); // deg >= 1 (self loops)
}

// znew = alpha*h0 + (1-alpha)*dinv[n]^2*z   (self-loop term folded in)
__global__ void appnp_init(float* __restrict__ znew, const float* __restrict__ z,
                           const float* __restrict__ h0, const float* __restrict__ dinv,
                           long n) {
    long i = blockIdx.x * (long)blockDim.x + threadIdx.x;
    if (i < n) {
        float di = dinv[i >> 7];
        znew[i] = ALPHA_F * h0[i] + (1.0f - ALPHA_F) * di * di * z[i];
    }
}

// znew[col, c] += (1-alpha)*dinv[row]*dinv[col]*z[row, c]
__global__ void appnp_edge(float* __restrict__ znew, const float* __restrict__ z,
                           const float* __restrict__ dinv,
                           const int* __restrict__ src, const int* __restrict__ dst,
                           int soff, int doff, int E) {
    long tid = blockIdx.x * (long)blockDim.x + threadIdx.x;
    if (tid >= (long)E * 32) return;
    int e  = (int)(tid >> 5);
    int c0 = (int)(tid & 31) << 2;
    int r = src[e] + soff;
    int c = dst[e] + doff;
    float w = (1.0f - ALPHA_F) * dinv[r] * dinv[c];
    const float4 v = *(const float4*)(z + (size_t)r * HID + c0);
    float* o = znew + (size_t)c * HID + c0;
    atomicAdd(o + 0, w * v.x);
    atomicAdd(o + 1, w * v.y);
    atomicAdd(o + 2, w * v.z);
    atomicAdd(o + 3, w * v.w);
}

// ---------------------------------------------------------------------------
// Orchestration
// ---------------------------------------------------------------------------
extern "C" void kernel_launch(void* const* d_in, const int* in_sizes, int n_in,
                              void* d_out, int out_size, void* d_ws, size_t ws_size,
                              hipStream_t stream) {
    (void)in_sizes; (void)n_in; (void)out_size; (void)ws_size;

    const float* x_gene = (const float*)d_in[0];
    const float* x_path = (const float*)d_in[1];
    const float* x_vacc = (const float*)d_in[2];
    const float* win_g  = (const float*)d_in[3];
    const float* bin_g  = (const float*)d_in[4];
    const float* win_p  = (const float*)d_in[5];
    const float* bin_p  = (const float*)d_in[6];
    const float* win_v  = (const float*)d_in[7];
    const float* bin_v  = (const float*)d_in[8];
    const float* wsl    = (const float*)d_in[9];   // [2,4,128,128]
    const float* bsl    = (const float*)d_in[10];  // [2,4,128]
    const float* wsr    = (const float*)d_in[11];  // [2,4,128,128]
    const int* ei_tg = (const int*)d_in[12];       // vaccine -> gene
    const int* ei_in = (const int*)d_in[13];       // gene -> pathway
    const int* ei_ct = (const int*)d_in[14];       // pathway -> gene
    const int* ei_ii = (const int*)d_in[15];       // gene -> gene

    // Workspace layout (f32): ~83 MB
    float* h0   = (float*)d_ws;                  // [NT,128]  node state / APPNP teleport
    float* zA   = h0 + (size_t)NT * HID;         // [NT,128]  o_gene/o_pathway, APPNP ping
    float* deg  = zA + (size_t)NT * HID;         // [NT]      per-relation cnt / APPNP degree
    float* dinv = deg + NT;                      // [NT]
    float* hg = h0;
    float* hp = h0 + (size_t)NG * HID;
    float* hv = h0 + (size_t)(NG + NP) * HID;
    float* og = zA;                              // [NG,128]
    float* op = zA + (size_t)NG * HID;           // [NP,128]
    float* dout = (float*)d_out;
    float* agg  = dout;                          // SAGE agg scratch (<= NT*HID)

    const int B = 256;
    auto nb = [](long n, int b) { return (unsigned)((n + b - 1) / b); };

    // ---- Phase 1: input projections + ReLU (WMMA) ----
    gemm_nt_wmma<256, true, true, false><<<dim3((NG + 63) / 64), 128, 0, stream>>>(
        x_gene, win_g, bin_g, hg, NG);
    gemm_nt_wmma<128, true, true, false><<<dim3((NP + 63) / 64), 128, 0, stream>>>(
        x_path, win_p, bin_p, hp, NP);
    gemm_nt_wmma<64, true, true, false><<<dim3((NV + 63) / 64), 128, 0, stream>>>(
        x_vacc, win_v, bin_v, hv, NV);

    // ---- Phase 2: hetero-SAGE layers ----
    auto relation = [&](const float* srcf, const int* ei, int ndst,
                        const float* wl, const float* bl, const float* wr,
                        const float* xdst, float* oacc) {
        const int* s = ei;
        const int* d = ei + EDG;
        fill_k<<<nb((long)ndst * HID, B), B, 0, stream>>>(agg, 0.0f, (long)ndst * HID);
        fill_k<<<nb(ndst, B), B, 0, stream>>>(deg, 0.0f, ndst);
        scatter_feat<<<nb((long)EDG * 32, B), B, 0, stream>>>(srcf, s, d, agg, EDG);
        count_edges<<<nb(EDG, B), B, 0, stream>>>(d, deg, EDG, 0);
        mean_div<<<nb((long)ndst * HID, B), B, 0, stream>>>(agg, deg, (long)ndst * HID);
        gemm_nt_wmma<128, true, false, true><<<dim3((ndst + 63) / 64), 128, 0, stream>>>(
            agg, wl, bl, oacc, ndst);
        gemm_nt_wmma<128, false, false, true><<<dim3((ndst + 63) / 64), 128, 0, stream>>>(
            xdst, wr, nullptr, oacc, ndst);
    };

    for (int l = 0; l < 2; ++l) {
        const float* WL = wsl + (size_t)l * 4 * HID * HID;
        const float* BL = bsl + (size_t)l * 4 * HID;
        const float* WR = wsr + (size_t)l * 4 * HID * HID;
        fill_k<<<nb((long)NG * HID, B), B, 0, stream>>>(og, 0.0f, (long)NG * HID);
        fill_k<<<nb((long)NP * HID, B), B, 0, stream>>>(op, 0.0f, (long)NP * HID);
        relation(hv, ei_tg, NG, WL + 0 * HID * HID, BL + 0 * HID, WR + 0 * HID * HID, hg, og);
        relation(hg, ei_in, NP, WL + 1 * HID * HID, BL + 1 * HID, WR + 1 * HID * HID, hp, op);
        relation(hp, ei_ct, NG, WL + 2 * HID * HID, BL + 2 * HID, WR + 2 * HID * HID, hg, og);
        relation(hg, ei_ii, NG, WL + 3 * HID * HID, BL + 3 * HID, WR + 3 * HID * HID, hg, og);
        combine_scale_relu<<<nb((long)NG * HID, B), B, 0, stream>>>(hg, og, 1.0f / 3.0f,
                                                                    (long)NG * HID);
        combine_scale_relu<<<nb((long)NP * HID, B), B, 0, stream>>>(hp, op, 1.0f,
                                                                    (long)NP * HID);
        relu_ip<<<nb((long)NV * HID, B), B, 0, stream>>>(hv, (long)NV * HID);
    }

    // ---- Phase 3: APPNP ----
    // deg: 1 (self loop) + in-degree over combined graph columns
    fill_k<<<nb(NT, B), B, 0, stream>>>(deg, 1.0f, NT);
    count_edges<<<nb(EDG, B), B, 0, stream>>>(ei_tg + EDG, deg, EDG, 0);    // -> gene
    count_edges<<<nb(EDG, B), B, 0, stream>>>(ei_in + EDG, deg, EDG, NG);   // -> pathway
    count_edges<<<nb(EDG, B), B, 0, stream>>>(ei_ct + EDG, deg, EDG, 0);    // -> gene
    count_edges<<<nb(EDG, B), B, 0, stream>>>(ei_ii + EDG, deg, EDG, 0);    // -> gene
    rsqrt_k<<<nb(NT, B), B, 0, stream>>>(dinv, deg, NT);

    const float* z = h0;
    for (int k = 0; k < K_APPNP; ++k) {
        float* zn = (k & 1) ? dout : zA;   // iter 7 (last) lands in d_out
        appnp_init<<<nb((long)NT * HID, B), B, 0, stream>>>(zn, z, h0, dinv,
                                                            (long)NT * HID);
        appnp_edge<<<nb((long)EDG * 32, B), B, 0, stream>>>(zn, z, dinv, ei_tg,
                                                            ei_tg + EDG, NG + NP, 0, EDG);
        appnp_edge<<<nb((long)EDG * 32, B), B, 0, stream>>>(zn, z, dinv, ei_in,
                                                            ei_in + EDG, 0, NG, EDG);
        appnp_edge<<<nb((long)EDG * 32, B), B, 0, stream>>>(zn, z, dinv, ei_ct,
                                                            ei_ct + EDG, NG, 0, EDG);
        appnp_edge<<<nb((long)EDG * 32, B), B, 0, stream>>>(zn, z, dinv, ei_ii,
                                                            ei_ii + EDG, 0, 0, EDG);
        z = zn;
    }
}
